// NeuronLlama4VisionMLP_80376017977935
// MI455X (gfx1250) — matrix operand order repaired and checked by reading, not compiled, not run on hardware
//
#include <hip/hip_runtime.h>
#include <hip/hip_bf16.h>
#include <math.h>

// ---------------------------------------------------------------------------
// Problem sizes (compile-time)
// ---------------------------------------------------------------------------
#define MM   16384      // B*S
#define HH   1408       // hidden
#define II   5632       // intermediate
#define EPS  1e-6f

typedef __attribute__((ext_vector_type(16))) _Float16 v16h;
typedef __attribute__((ext_vector_type(8)))  _Float16 v8h;
typedef __attribute__((ext_vector_type(4)))  _Float16 v4h;
typedef __attribute__((ext_vector_type(8)))  float    v8f;

union V16 { v16h v; v8h h[2]; };

// ---------------------------------------------------------------------------
// Kernel 1: tiled f32 -> f16 transpose.  out[c*R + r] = (f16) in[r*C + c]
// grid = (C/32, R/32), block = 256
// ---------------------------------------------------------------------------
__global__ __launch_bounds__(256)
void transpose_f32_to_f16(const float* __restrict__ in, _Float16* __restrict__ out,
                          int R, int C) {
  __shared__ float tile[32][33];
  const int tx = threadIdx.x & 31;
  const int ty = threadIdx.x >> 5;            // 0..7
  const int rBase = blockIdx.y * 32;
  const int cBase = blockIdx.x * 32;
#pragma unroll
  for (int j = 0; j < 4; ++j) {
    int r = ty + j * 8;
    tile[r][tx] = in[(size_t)(rBase + r) * C + (cBase + tx)];
  }
  __syncthreads();
#pragma unroll
  for (int j = 0; j < 4; ++j) {
    int c = ty + j * 8;                       // output row within tile = input col
    out[(size_t)(cBase + c) * R + (rBase + tx)] = (_Float16)tile[tx][c];
  }
}

// ---------------------------------------------------------------------------
// Kernel 2: fused residual add + RMSNorm.
//   h = x + residual          (f32, written to hOut — second reference output)
//   normed = h * rsqrt(mean(h^2)+eps) * w   (f16, to workspace)
// One block (256 threads, 8 waves) per row of length HH = 1408 = 352 float4.
// ---------------------------------------------------------------------------
__global__ __launch_bounds__(256)
void fused_add_rmsnorm(const float* __restrict__ x, const float* __restrict__ res,
                       const float* __restrict__ w,
                       float* __restrict__ hOut, _Float16* __restrict__ normed) {
  const int row  = blockIdx.x;
  const int t    = threadIdx.x;
  const int lane = t & 31;
  const int wave = t >> 5;

  const float4* x4 = (const float4*)(x   + (size_t)row * HH);
  const float4* r4 = (const float4*)(res + (size_t)row * HH);
  float4*       h4 = (float4*)(hOut + (size_t)row * HH);

  // 352 float4 per row: every thread does i=t, threads t<96 also do i=256+t.
  float4 a = x4[t], b = r4[t];
  float4 h0 = make_float4(a.x + b.x, a.y + b.y, a.z + b.z, a.w + b.w);
  h4[t] = h0;
  float ss = h0.x * h0.x + h0.y * h0.y + h0.z * h0.z + h0.w * h0.w;

  float4 h1 = make_float4(0.f, 0.f, 0.f, 0.f);
  if (t < 96) {
    float4 a1 = x4[256 + t], b1 = r4[256 + t];
    h1 = make_float4(a1.x + b1.x, a1.y + b1.y, a1.z + b1.z, a1.w + b1.w);
    h4[256 + t] = h1;
    ss += h1.x * h1.x + h1.y * h1.y + h1.z * h1.z + h1.w * h1.w;
  }

  // wave32 reduce, then cross-wave via LDS
#pragma unroll
  for (int off = 16; off > 0; off >>= 1) ss += __shfl_down(ss, off, 32);
  __shared__ float wsum[8];
  __shared__ float srms;
  if (lane == 0) wsum[wave] = ss;
  __syncthreads();
  if (t == 0) {
    float tot = 0.f;
#pragma unroll
    for (int i = 0; i < 8; ++i) tot += wsum[i];
    srms = rsqrtf(tot / (float)HH + EPS);
  }
  __syncthreads();
  const float rms = srms;

  const float4* w4 = (const float4*)w;
  v4h* n4 = (v4h*)(normed + (size_t)row * HH);
  {
    float4 g = w4[t];
    v4h o; o[0] = (_Float16)(h0.x * rms * g.x); o[1] = (_Float16)(h0.y * rms * g.y);
           o[2] = (_Float16)(h0.z * rms * g.z); o[3] = (_Float16)(h0.w * rms * g.w);
    n4[t] = o;
  }
  if (t < 96) {
    float4 g = w4[256 + t];
    v4h o; o[0] = (_Float16)(h1.x * rms * g.x); o[1] = (_Float16)(h1.y * rms * g.y);
           o[2] = (_Float16)(h1.z * rms * g.z); o[3] = (_Float16)(h1.w * rms * g.w);
    n4[256 + t] = o;
  }
}

// ---------------------------------------------------------------------------
// Kernel 3: WMMA GEMM.  C[M,N] = A[M,K] * BT[N,K]^T  (+bias, optional GELU)
//   A, BT are f16 row-major (BT is the pre-transposed weight: K contiguous).
//   GELU_F16: epilogue = erf-GELU -> f16 store (activation to workspace)
//   else:     epilogue = +bias    -> f32 store (final output)
// Block tile 128x128, K-tile 64. 256 threads = 8 waves (2 M x 4 N),
// wave tile 64x32 = 4x2 fragments of v_wmma_f32_16x16x32_f16.
//
// Tile staging: thread t owns the same 16-byte K-chunk (k8 = (t&7)*8 halves)
// at four rows lrow, lrow+32, lrow+64, lrow+96 — 8 named float4 registers,
// no arrays/allocas, so nothing can be demoted to scratch.
// ---------------------------------------------------------------------------
#define BM  128
#define BN  128
#define BK  64
#define LDT 72    // padded halves per LDS row (144 B: 16B-aligned, conflict-free)

template <bool GELU_F16>
__global__ __launch_bounds__(256)
void mlp_gemm(const _Float16* __restrict__ A, const _Float16* __restrict__ BT,
              const float* __restrict__ bias, void* __restrict__ Cout,
              int M, int N, int K) {
  __shared__ _Float16 As[BM * LDT];
  __shared__ _Float16 Bs[BN * LDT];

  const int t    = threadIdx.x;
  const int lane = t & 31;
  const int half = lane >> 4;     // 0/1
  const int l15  = lane & 15;
  const int wave = t >> 5;
  const int waveM = (wave >> 2) * 64;   // 0 / 64
  const int waveN = (wave & 3)  * 32;   // 0..96

  const int mBase = blockIdx.y * BM;
  const int nBase = blockIdx.x * BN;

  // per-thread staging coordinates
  const int lrow = t >> 3;              // 0..31
  const int k8   = (t & 7) * 8;         // halves: 0..56
  const size_t rowStep = (size_t)32 * K;
  const int    sOff    = lrow * LDT + k8;

  const _Float16* __restrict__ pA = A  + (size_t)(mBase + lrow) * K + k8;
  const _Float16* __restrict__ pB = BT + (size_t)(nBase + lrow) * K + k8;
  _Float16* __restrict__ sA = As + sOff;
  _Float16* __restrict__ sB = Bs + sOff;

  v8f acc[4][2] = {};

  // first tile prefetch (8 named float4 registers)
  float4 a0 = *(const float4*)(pA);
  float4 a1 = *(const float4*)(pA + rowStep);
  float4 a2 = *(const float4*)(pA + 2 * rowStep);
  float4 a3 = *(const float4*)(pA + 3 * rowStep);
  float4 b0 = *(const float4*)(pB);
  float4 b1 = *(const float4*)(pB + rowStep);
  float4 b2 = *(const float4*)(pB + 2 * rowStep);
  float4 b3 = *(const float4*)(pB + 3 * rowStep);
  pA += BK; pB += BK;

  const int nk = K / BK;
  for (int kt = 0; kt < nk; ++kt) {
    __syncthreads();
    *(float4*)(sA)                = a0;
    *(float4*)(sA + 32 * LDT)     = a1;
    *(float4*)(sA + 64 * LDT)     = a2;
    *(float4*)(sA + 96 * LDT)     = a3;
    *(float4*)(sB)                = b0;
    *(float4*)(sB + 32 * LDT)     = b1;
    *(float4*)(sB + 64 * LDT)     = b2;
    *(float4*)(sB + 96 * LDT)     = b3;
    __syncthreads();
    if (kt + 1 < nk) {
      a0 = *(const float4*)(pA);
      a1 = *(const float4*)(pA + rowStep);
      a2 = *(const float4*)(pA + 2 * rowStep);
      a3 = *(const float4*)(pA + 3 * rowStep);
      b0 = *(const float4*)(pB);
      b1 = *(const float4*)(pB + rowStep);
      b2 = *(const float4*)(pB + 2 * rowStep);
      b3 = *(const float4*)(pB + 3 * rowStep);
      pA += BK; pB += BK;
    }
#pragma unroll
    for (int kk = 0; kk < BK; kk += 32) {
      V16 af[4], bf[2];
#pragma unroll
      for (int mi = 0; mi < 4; ++mi) {
        const _Float16* p = &As[(waveM + mi * 16 + l15) * LDT + kk + 8 * half];
        af[mi].h[0] = *(const v8h*)(p);        // K chunk 0 (8 halves)
        af[mi].h[1] = *(const v8h*)(p + 16);   // K chunk 1
      }
#pragma unroll
      for (int ni = 0; ni < 2; ++ni) {
        const _Float16* p = &Bs[(waveN + ni * 16 + l15) * LDT + kk + 8 * half];
        bf[ni].h[0] = *(const v8h*)(p);        // K chunk 0 (8 halves)
        bf[ni].h[1] = *(const v8h*)(p + 16);   // K chunk 1
      }
#pragma unroll
      for (int mi = 0; mi < 4; ++mi)
#pragma unroll
        for (int ni = 0; ni < 2; ++ni)
          acc[mi][ni] = __builtin_amdgcn_wmma_f32_16x16x32_f16(
              false, af[mi].v, false, bf[ni].v, (short)0, acc[mi][ni],
              false, false);
    }
  }

  // epilogue: C layout — VGPR v: M = v + 8*half, N = l15 (per 16x16 fragment)
#pragma unroll
  for (int mi = 0; mi < 4; ++mi) {
#pragma unroll
    for (int ni = 0; ni < 2; ++ni) {
      const int n  = nBase + waveN + ni * 16 + l15;
      const float bv = bias[n];
#pragma unroll
      for (int v = 0; v < 8; ++v) {
        const int m = mBase + waveM + mi * 16 + half * 8 + v;
        float val = acc[mi][ni][v] + bv;
        if (GELU_F16) {
          val = 0.5f * val * (1.0f + erff(val * 0.70710678118654752f));
          ((_Float16*)Cout)[(size_t)m * N + n] = (_Float16)val;
        } else {
          ((float*)Cout)[(size_t)m * N + n] = val;
        }
      }
    }
  }
}

// ---------------------------------------------------------------------------
// Launch.  d_in: x, residual, ln_w, W_up, b_up, W_down, b_down (all f32).
// d_out: [ out (M*H f32) | h (M*H f32) ].
// Workspace layout (f16):
//   normed  M*H   (46.1 MB)
//   W_upT   I*H   (15.9 MB)   row-major [I][H]
//   W_dnT   H*I   (15.9 MB)   row-major [H][I]
//   act     M*I   (184.5 MB)
// total ~262.4 MB
// ---------------------------------------------------------------------------
extern "C" void kernel_launch(void* const* d_in, const int* in_sizes, int n_in,
                              void* d_out, int out_size, void* d_ws, size_t ws_size,
                              hipStream_t stream) {
  (void)in_sizes; (void)n_in; (void)out_size; (void)ws_size;

  const float* x     = (const float*)d_in[0];
  const float* resid = (const float*)d_in[1];
  const float* ln_w  = (const float*)d_in[2];
  const float* W_up  = (const float*)d_in[3];
  const float* b_up  = (const float*)d_in[4];
  const float* W_dn  = (const float*)d_in[5];
  const float* b_dn  = (const float*)d_in[6];

  float* out  = (float*)d_out;
  float* hOut = out + (size_t)MM * HH;

  _Float16* normed = (_Float16*)d_ws;
  _Float16* wupT   = normed + (size_t)MM * HH;
  _Float16* wdnT   = wupT   + (size_t)II * HH;
  _Float16* act    = wdnT   + (size_t)HH * II;

  // 1) weight conversion + transpose (f32 -> f16, K-contiguous for B operand)
  transpose_f32_to_f16<<<dim3(II / 32, HH / 32), 256, 0, stream>>>(W_up, wupT, HH, II);
  transpose_f32_to_f16<<<dim3(HH / 32, II / 32), 256, 0, stream>>>(W_dn, wdnT, II, HH);

  // 2) h = x + residual ; normed = RMSNorm(h)
  fused_add_rmsnorm<<<MM, 256, 0, stream>>>(x, resid, ln_w, hOut, normed);

  // 3) act = gelu(normed @ W_up + b_up)    [M,I] f16
  mlp_gemm<true><<<dim3(II / BN, MM / BM), 256, 0, stream>>>(
      normed, wupT, b_up, act, MM, II, HH);

  // 4) out = act @ W_down + b_down         [M,H] f32
  mlp_gemm<false><<<dim3(HH / BN, MM / BM), 256, 0, stream>>>(
      act, wdnT, b_dn, out, MM, HH, II);
}
